// GraphTransformer_35966056137280
// MI455X (gfx1250) — matrix operand order repaired and checked by reading, not compiled, not run on hardware
//
#include <hip/hip_runtime.h>
#include <hip/hip_bf16.h>
#include <cstdint>

typedef __attribute__((ext_vector_type(16))) _Float16 v16h;
typedef __attribute__((ext_vector_type(8)))  float    v8f;

#define NB 8
#define NN 1024
#define MM 1025
#define DIMX 128
#define NHEADS 8
#define TOK (NB*MM)          /* 8200 */
#define EPSLN 1e-5f
#define ATTN_SCALE 0.08838834764831845f
#define USE_ASYNC_LDS 1

union Frag16 { v16h v; float4 f4[2]; };
union H8 { float4 f4; _Float16 h[8]; };

static __device__ __forceinline__ float gelu_f(float x) {
    return 0.5f * x * (1.0f + erff(x * 0.70710678118654752f));
}
static __device__ __forceinline__ v8f wmma16(v16h a, v16h b, v8f c) {
    return __builtin_amdgcn_wmma_f32_16x16x32_f16(false, a, false, b, (short)0, c, false, false);
}
static __device__ __forceinline__ float wave_sum32(float v) {
#pragma unroll
    for (int m = 16; m >= 1; m >>= 1) v += __shfl_xor(v, m, 32);
    return v;
}
static __device__ __forceinline__ float half_sum16(float v) {
#pragma unroll
    for (int m = 8; m >= 1; m >>= 1) v += __shfl_xor(v, m, 32);
    return v;
}
static __device__ __forceinline__ float half_max16(float v) {
#pragma unroll
    for (int m = 8; m >= 1; m >>= 1) v = fmaxf(v, __shfl_xor(v, m, 32));
    return v;
}
// 16B global -> LDS copy; ASYNCcnt-tracked on the CDNA5 path.
static __device__ __forceinline__ void async_copy_b128(void* lds, const void* g) {
#if USE_ASYNC_LDS
    uint32_t l = (uint32_t)(uintptr_t)lds;   // low 32 bits of generic ptr = LDS byte addr
    unsigned long long ga = (unsigned long long)(uintptr_t)g;
    asm volatile("global_load_async_to_lds_b128 %0, %1, off" :: "v"(l), "v"(ga) : "memory");
#else
    *(float4*)lds = *(const float4*)g;
#endif
}
static __device__ __forceinline__ void async_wait() {
#if USE_ASYNC_LDS
    asm volatile("s_wait_asynccnt 0x0" ::: "memory");
#endif
}

// ---------------------------------------------------------------- converts
__global__ void cvt_f32_f16(const float* __restrict__ s, _Float16* __restrict__ d, int n) {
    int i = blockIdx.x * blockDim.x + threadIdx.x;
    if (i < n) d[i] = (_Float16)s[i];
}

__global__ void init_cls(const float* __restrict__ cls, float* __restrict__ xF) {
    int t = blockIdx.x * blockDim.x + threadIdx.x;
    if (t < NB * DIMX) {
        int b = t / DIMX, j = t % DIMX;
        xF[(size_t)(b * MM) * DIMX + j] = cls[j];
    }
}

// ---------------------------------------------------------------- PE MLP (35->70->70->70->LN->128)
__global__ void pe_kernel(const float* __restrict__ nf, const float* __restrict__ lp,
                          const float* __restrict__ w1, const float* __restrict__ b1,
                          const float* __restrict__ w2, const float* __restrict__ b2,
                          const float* __restrict__ w3, const float* __restrict__ b3,
                          const float* __restrict__ lng, const float* __restrict__ lnb,
                          const float* __restrict__ w4, const float* __restrict__ b4,
                          float* __restrict__ xF) {
    int t = blockIdx.x * blockDim.x + threadIdx.x;
    if (t >= NB * NN) return;
    int b = t / NN, n = t % NN;
    float in[35];
    for (int i = 0; i < 3; ++i)  in[i]     = nf[(size_t)(b * NN + n) * 3 + i];
    for (int i = 0; i < 32; ++i) in[3 + i] = lp[(size_t)(b * NN + n) * 32 + i];
    float a[70], c[70];
    for (int j = 0; j < 70; ++j) {
        float s = b1[j];
        for (int i = 0; i < 35; ++i) s += in[i] * w1[i * 70 + j];
        a[j] = gelu_f(s);
    }
    for (int j = 0; j < 70; ++j) {
        float s = b2[j];
        for (int i = 0; i < 70; ++i) s += a[i] * w2[i * 70 + j];
        c[j] = gelu_f(s);
    }
    for (int j = 0; j < 70; ++j) {
        float s = b3[j];
        for (int i = 0; i < 70; ++i) s += c[i] * w3[i * 70 + j];
        a[j] = gelu_f(s);
    }
    float m = 0.f;
    for (int j = 0; j < 70; ++j) m += a[j];
    m *= (1.f / 70.f);
    float v = 0.f;
    for (int j = 0; j < 70; ++j) { float d = a[j] - m; v += d * d; }
    float rs = rsqrtf(v * (1.f / 70.f) + EPSLN);
    for (int j = 0; j < 70; ++j) c[j] = (a[j] - m) * rs * lng[j] + lnb[j];
    size_t row = (size_t)(b * MM + 1 + n) * DIMX;
    for (int j = 0; j < DIMX; ++j) {
        float s = b4[j];
        for (int i = 0; i < 70; ++i) s += c[i] * w4[i * DIMX + j];
        xF[row + j] = s;
    }
}

// ---------------------------------------------------------------- LayerNorm (+optional gamma gate), wave per row
__global__ void ln_kernel(const float* __restrict__ x, const float* __restrict__ g,
                          const float* __restrict__ bb, float* __restrict__ hF,
                          _Float16* __restrict__ h16, const float* __restrict__ gw,
                          const float* __restrict__ gb, float* __restrict__ gamma, int rows) {
    int row = blockIdx.x * 8 + (threadIdx.x >> 5);
    int lane = threadIdx.x & 31;
    if (row >= rows) return;
    size_t base = (size_t)row * DIMX;
    float v[4];
#pragma unroll
    for (int k = 0; k < 4; ++k) v[k] = x[base + lane + 32 * k];
    float s = wave_sum32(v[0] + v[1] + v[2] + v[3]);
    float mean = s * (1.f / 128.f);
    float vs = 0.f;
#pragma unroll
    for (int k = 0; k < 4; ++k) { float d = v[k] - mean; vs += d * d; }
    vs = wave_sum32(vs);
    float rsd = rsqrtf(vs * (1.f / 128.f) + EPSLN);
    float h[4];
    float g0 = 0.f, g1 = 0.f;
#pragma unroll
    for (int k = 0; k < 4; ++k) {
        int c = lane + 32 * k;
        h[k] = (v[k] - mean) * rsd * g[c] + bb[c];
        hF[base + c] = h[k];
        h16[base + c] = (_Float16)h[k];
        if (gw) { g0 += h[k] * gw[c * 2]; g1 += h[k] * gw[c * 2 + 1]; }
    }
    if (gw) {
        g0 = wave_sum32(g0); g1 = wave_sum32(g1);
        if (lane == 0) {
            gamma[(size_t)row * 2]     = __expf(g0 + gb[0]);
            gamma[(size_t)row * 2 + 1] = __expf(g1 + gb[1]);
        }
    }
}

// ---------------------------------------------------------------- generic WMMA GEMM, 64x64 block tile, 8 waves
// each wave: one 16-row tile, two 16-col tiles (A-frag reused, 2 WMMAs/K-step)
// mode 0: QKV scatter -> Qo/Ko/Vo (f16, [B,H,M,128])
// mode 1: outF = acc + bias + resid (f32)
// mode 2: outH = gelu(acc + bias) (f16)
__global__ void gemm_kernel(const _Float16* __restrict__ A, const _Float16* __restrict__ Bw,
                            int Mr, int K, int N, int mode,
                            const float* __restrict__ bias, const float* __restrict__ resid,
                            float* __restrict__ outF, _Float16* __restrict__ outH,
                            _Float16* __restrict__ Qo, _Float16* __restrict__ Ko, _Float16* __restrict__ Vo) {
    __shared__ __align__(16) _Float16 As[64][40];   // row-major rows x k
    __shared__ __align__(16) _Float16 Bt[64][40];   // transposed: [n][k]
    int t = threadIdx.x;
    int m0 = blockIdx.x * 64;
    int n0 = blockIdx.y * 64;
    int wid = t >> 5, lane = t & 31;
    int wm = wid & 3, wn = wid >> 2;          // 4 row-tiles x 2 col-groups (of 32)
    int r = lane & 15, sel = lane >> 4;
    v8f acc0 = {0.f, 0.f, 0.f, 0.f, 0.f, 0.f, 0.f, 0.f};
    v8f acc1 = {0.f, 0.f, 0.f, 0.f, 0.f, 0.f, 0.f, 0.f};

    int ar = t >> 2, ac = (t & 3) * 8;         // A: 64 rows x 32 cols, 16B per thread
    int gr = m0 + ar; if (gr >= Mr) gr = Mr - 1;   // clamp: dup row, C rows discarded at store
    int br = t >> 3, bc = (t & 7) * 8;         // B: 32 k-rows x 64 n-cols, 16B per thread

    for (int kk = 0; kk < K; kk += 32) {
        async_copy_b128(&As[ar][ac], A + (size_t)gr * K + kk + ac);
        {
            H8 bv; bv.f4 = *(const float4*)(Bw + (size_t)(kk + br) * N + n0 + bc);
#pragma unroll
            for (int e = 0; e < 8; ++e) Bt[bc + e][br] = bv.h[e];
        }
        __builtin_prefetch(A + (size_t)gr * K + kk + 32 + ac, 0, 1);
        async_wait();
        __syncthreads();
        Frag16 fa, fb0, fb1;
        {
            const _Float16* pa = &As[wm * 16 + r][sel ? 8 : 0];
            fa.f4[0] = *(const float4*)pa;
            fa.f4[1] = *(const float4*)(pa + 16);
            const _Float16* pb0 = &Bt[wn * 32 + r][sel * 8];
            fb0.f4[0] = *(const float4*)pb0;
            fb0.f4[1] = *(const float4*)(pb0 + 16);
            const _Float16* pb1 = &Bt[wn * 32 + 16 + r][sel * 8];
            fb1.f4[0] = *(const float4*)pb1;
            fb1.f4[1] = *(const float4*)(pb1 + 16);
        }
        acc0 = wmma16(fa.v, fb0.v, acc0);
        acc1 = wmma16(fa.v, fb1.v, acc1);
        __syncthreads();
    }

#pragma unroll
    for (int half = 0; half < 2; ++half) {
        v8f acc = half ? acc1 : acc0;
        int cc = n0 + wn * 32 + half * 16 + r;
#pragma unroll
        for (int j = 0; j < 8; ++j) {
            int rr = m0 + wm * 16 + sel * 8 + j;
            if (rr >= Mr) continue;
            float val = acc[j];
            if (mode == 0) {
                int bb = rr / MM, mt = rr % MM;
                int s3 = cc >> 10, hh = (cc >> 7) & 7, dd = cc & 127;
                size_t di = ((size_t)(bb * NHEADS + hh) * MM + mt) * DIMX + dd;
                _Float16 hv = (_Float16)val;
                if (s3 == 0) Qo[di] = hv; else if (s3 == 1) Ko[di] = hv; else Vo[di] = hv;
            } else if (mode == 1) {
                outF[(size_t)rr * N + cc] = val + bias[cc] + resid[(size_t)rr * N + cc];
            } else {
                outH[(size_t)rr * N + cc] = (_Float16)gelu_f(val + bias[cc]);
            }
        }
    }
}

// ---------------------------------------------------------------- flash attention w/ gamma gate + adjacency bias
static __device__ __forceinline__ float adjv(const float* __restrict__ adj, int b, int q, int ky) {
    if (q == 0) return (ky == 0) ? 1.f : 0.f;
    if (ky == 0) return 0.f;
    return adj[((size_t)b * NN + (q - 1)) * NN + (ky - 1)];
}

__global__ void attn_kernel(const _Float16* __restrict__ Q, const _Float16* __restrict__ Km,
                            const _Float16* __restrict__ V, const float* __restrict__ gamma,
                            const float* __restrict__ adj, _Float16* __restrict__ O) {
    __shared__ __align__(16) _Float16 Ks[32][136];      // [key][dim], shared by all 8 waves
    __shared__ __align__(16) _Float16 Vt[128][40];      // [dim][key], transposed
    __shared__ __align__(16) _Float16 Pl[8][16 * 40];   // per-wave P transpose scratch
    int bh = blockIdx.x;
    int b = bh >> 3, h = bh & 7;
    int t = threadIdx.x;
    int wid = t >> 5, lane = t & 31;
    int qt = blockIdx.y * 8 + wid;               // query tile (0..71, 65 valid)
    int r = lane & 15, sel = lane >> 4;
    const _Float16* Qb = Q + (size_t)bh * MM * DIMX;
    const _Float16* Kb = Km + (size_t)bh * MM * DIMX;
    const _Float16* Vb = V + (size_t)bh * MM * DIMX;

    // gamma gate per output row (rows sel*8+j)
    float g0s[8], g1s[8];
#pragma unroll
    for (int j = 0; j < 8; ++j) {
        int q = qt * 16 + sel * 8 + j;
        if (q < MM) {
            g0s[j] = gamma[(size_t)(b * MM + q) * 2] * ATTN_SCALE;
            g1s[j] = gamma[(size_t)(b * MM + q) * 2 + 1];
        } else { g0s[j] = 0.f; g1s[j] = 0.f; }
    }
    // Q fragments: 16x128 = 4 chunks of 16x32, two 16B loads each
    int qa = qt * 16 + r;
    Frag16 qf[4];
#pragma unroll
    for (int kc = 0; kc < 4; ++kc) {
        if (qa < MM) {
            const _Float16* p = Qb + (size_t)qa * DIMX + kc * 32 + (sel ? 8 : 0);
            qf[kc].f4[0] = *(const float4*)p;
            qf[kc].f4[1] = *(const float4*)(p + 16);
        } else {
            float4 z; z.x = z.y = z.z = z.w = 0.f;
            qf[kc].f4[0] = z; qf[kc].f4[1] = z;
        }
    }
    v8f o[8];
#pragma unroll
    for (int d = 0; d < 8; ++d)
#pragma unroll
        for (int j = 0; j < 8; ++j) o[d][j] = 0.f;
    float mrow[8], lrow[8];
#pragma unroll
    for (int j = 0; j < 8; ++j) { mrow[j] = -3e38f; lrow[j] = 0.f; }

    // staging index math (uniform work split across 256 threads)
    int krow0 = t >> 4, kch0 = t & 15;          // K: 2 chunks/thread of (row, 8-half chunk)
    int vrow = t >> 3, vch = t & 7;             // V: 16 halves/thread -> transposed scatter

    for (int it = 0; it < 33; ++it) {            // 33 * 32 = 1056 >= 1025 keys
        int kb = it * 32;
        __syncthreads();                          // protect previous tiles/Pl from overwrite
        // ---- stage K tile (32x128) via async copies, clamped rows
        {
            int ky0 = kb + krow0;       if (ky0 > NN) ky0 = NN;
            int ky1 = kb + krow0 + 16;  if (ky1 > NN) ky1 = NN;
            async_copy_b128(&Ks[krow0][kch0 * 8],      Kb + (size_t)ky0 * DIMX + kch0 * 8);
            async_copy_b128(&Ks[krow0 + 16][kch0 * 8], Kb + (size_t)ky1 * DIMX + kch0 * 8);
        }
        // ---- stage V tile transposed (Vt[dim][key])
        {
            int vy = kb + vrow; if (vy > NN) vy = NN;
            const _Float16* pv = Vb + (size_t)vy * DIMX + vch * 16;
            H8 v0, v1; v0.f4 = *(const float4*)pv; v1.f4 = *(const float4*)(pv + 8);
#pragma unroll
            for (int e = 0; e < 8; ++e) Vt[vch * 16 + e][vrow] = v0.h[e];
#pragma unroll
            for (int e = 0; e < 8; ++e) Vt[vch * 16 + 8 + e][vrow] = v1.h[e];
        }
        async_wait();
        __syncthreads();

        v8f S[2];
#pragma unroll
        for (int t2 = 0; t2 < 2; ++t2) {
            v8f s = {0.f, 0.f, 0.f, 0.f, 0.f, 0.f, 0.f, 0.f};
            int ky = kb + t2 * 16 + r;           // key handled by this lane (B col)
#pragma unroll
            for (int kc = 0; kc < 4; ++kc) {
                Frag16 fk;
                const _Float16* p = &Ks[t2 * 16 + r][kc * 32 + sel * 8];
                fk.f4[0] = *(const float4*)p;
                fk.f4[1] = *(const float4*)(p + 16);
                s = wmma16(qf[kc].v, fk.v, s);
            }
#pragma unroll
            for (int j = 0; j < 8; ++j) {
                int q = qt * 16 + sel * 8 + j;
                if (q < MM && ky < MM)
                    s[j] = g0s[j] * s[j] + g1s[j] * adjv(adj, b, q, ky);
                else
                    s[j] = -3e38f;
            }
            S[t2] = s;
        }
        // online softmax update
        float alpha[8], mnew[8];
#pragma unroll
        for (int j = 0; j < 8; ++j) {
            float mx = half_max16(fmaxf(S[0][j], S[1][j]));
            mnew[j] = fmaxf(mrow[j], mx);
            alpha[j] = __expf(mrow[j] - mnew[j]);
            mrow[j] = mnew[j];
        }
#pragma unroll
        for (int j = 0; j < 8; ++j) {
            float p0 = __expf(S[0][j] - mnew[j]);
            float p1 = __expf(S[1][j] - mnew[j]);
            S[0][j] = p0; S[1][j] = p1;
            float rsum = half_sum16(p0 + p1);
            lrow[j] = lrow[j] * alpha[j] + rsum;
        }
#pragma unroll
        for (int d = 0; d < 8; ++d)
#pragma unroll
            for (int j = 0; j < 8; ++j) o[d][j] *= alpha[j];
        // transpose P (C-layout) -> A-layout through per-wave LDS
#pragma unroll
        for (int t2 = 0; t2 < 2; ++t2)
#pragma unroll
            for (int j = 0; j < 8; ++j)
                Pl[wid][(sel * 8 + j) * 40 + t2 * 16 + r] = (_Float16)S[t2][j];
        __syncthreads();
        Frag16 pf;
        {
            const _Float16* p = &Pl[wid][r * 40 + (sel ? 8 : 0)];
            pf.f4[0] = *(const float4*)p;
            pf.f4[1] = *(const float4*)(p + 16);
        }
        // O += P @ V   (V fragments from transposed LDS tile, contiguous)
#pragma unroll
        for (int d = 0; d < 8; ++d) {
            Frag16 fv;
            const _Float16* p = &Vt[d * 16 + r][sel * 8];
            fv.f4[0] = *(const float4*)p;
            fv.f4[1] = *(const float4*)(p + 16);
            o[d] = wmma16(pf.v, fv.v, o[d]);
        }
    }
    // normalize + store, token-major (B,M,H*128)
    float linv[8];
#pragma unroll
    for (int j = 0; j < 8; ++j) linv[j] = 1.f / lrow[j];
#pragma unroll
    for (int d = 0; d < 8; ++d)
#pragma unroll
        for (int j = 0; j < 8; ++j) {
            int q = qt * 16 + sel * 8 + j;
            if (q < MM) {
                float val = o[d][j] * linv[j];
                O[((size_t)(b * MM + q)) * (NHEADS * DIMX) + h * DIMX + d * 16 + r] = (_Float16)val;
            }
        }
}

// ---------------------------------------------------------------- head (tiny, fp32), wave per batch row
__global__ void head_kernel(const float* __restrict__ xF,
                            const float* __restrict__ hlng, const float* __restrict__ hlnb,
                            const float* __restrict__ hw, const float* __restrict__ hb,
                            const float* __restrict__ w1, const float* __restrict__ b1,
                            const float* __restrict__ w2, const float* __restrict__ b2,
                            const float* __restrict__ w3, const float* __restrict__ b3,
                            const float* __restrict__ plng, const float* __restrict__ plnb,
                            const float* __restrict__ w4, const float* __restrict__ b4,
                            float* __restrict__ out) {
    __shared__ float S0[8][128];
    __shared__ float S1[8][128];
    int lane = threadIdx.x & 31;
    int b = threadIdx.x >> 5;
    float v[4];
#pragma unroll
    for (int k = 0; k < 4; ++k) v[k] = xF[(size_t)(b * MM) * DIMX + lane + 32 * k];
    float s = wave_sum32(v[0] + v[1] + v[2] + v[3]);
    float mean = s * (1.f / 128.f);
    float vs = 0.f;
#pragma unroll
    for (int k = 0; k < 4; ++k) { float d = v[k] - mean; vs += d * d; }
    vs = wave_sum32(vs);
    float rsd = rsqrtf(vs * (1.f / 128.f) + EPSLN);
#pragma unroll
    for (int k = 0; k < 4; ++k) {
        int c = lane + 32 * k;
        S0[b][c] = (v[k] - mean) * rsd * hlng[c] + hlnb[c];
    }
    __syncthreads();
#pragma unroll
    for (int k = 0; k < 4; ++k) {
        int c = lane + 32 * k;
        float acc = hb[c];
        for (int i = 0; i < 128; ++i) acc += S0[b][i] * hw[i * 128 + c];
        S1[b][c] = acc;
        out[b * 128 + c] = acc;
    }
    __syncthreads();
#pragma unroll
    for (int k = 0; k < 4; ++k) {
        int c = lane + 32 * k;
        float acc = b1[c];
        for (int i = 0; i < 128; ++i) acc += S1[b][i] * w1[i * 128 + c];
        S0[b][c] = gelu_f(acc);
    }
    __syncthreads();
#pragma unroll
    for (int k = 0; k < 4; ++k) {
        int c = lane + 32 * k;
        float acc = b2[c];
        for (int i = 0; i < 128; ++i) acc += S0[b][i] * w2[i * 128 + c];
        S1[b][c] = gelu_f(acc);
    }
    __syncthreads();
    float p3[4];
#pragma unroll
    for (int k = 0; k < 4; ++k) {
        int c = lane + 32 * k;
        float acc = b3[c];
        for (int i = 0; i < 128; ++i) acc += S1[b][i] * w3[i * 128 + c];
        p3[k] = gelu_f(acc);
    }
    float ps = wave_sum32(p3[0] + p3[1] + p3[2] + p3[3]);
    float pm = ps * (1.f / 128.f);
    float pv = 0.f;
#pragma unroll
    for (int k = 0; k < 4; ++k) { float d = p3[k] - pm; pv += d * d; }
    pv = wave_sum32(pv);
    float prs = rsqrtf(pv * (1.f / 128.f) + EPSLN);
#pragma unroll
    for (int k = 0; k < 4; ++k) {
        int c = lane + 32 * k;
        S0[b][c] = (p3[k] - pm) * prs * plng[c] + plnb[c];
    }
    __syncthreads();
    for (int c = lane; c < 1000; c += 32) {
        float acc = b4[c];
        for (int i = 0; i < 128; ++i) acc += S0[b][i] * w4[i * 1000 + c];
        out[NB * 128 + b * 1000 + c] = acc;
    }
}

// ---------------------------------------------------------------- launch
extern "C" void kernel_launch(void* const* d_in, const int* in_sizes, int n_in,
                              void* d_out, int out_size, void* d_ws, size_t ws_size,
                              hipStream_t stream) {
    (void)in_sizes; (void)n_in; (void)out_size; (void)ws_size;
    const float* node_feat = (const float*)d_in[0];
    const float* adj       = (const float*)d_in[1];
    const float* lapl      = (const float*)d_in[2];
    const float* cls       = (const float*)d_in[3];
    const float* pe_w1 = (const float*)d_in[4];  const float* pe_b1 = (const float*)d_in[5];
    const float* pe_w2 = (const float*)d_in[6];  const float* pe_b2 = (const float*)d_in[7];
    const float* pe_w3 = (const float*)d_in[8];  const float* pe_b3 = (const float*)d_in[9];
    const float* pe_lng = (const float*)d_in[10]; const float* pe_lnb = (const float*)d_in[11];
    const float* pe_w4 = (const float*)d_in[12]; const float* pe_b4 = (const float*)d_in[13];
    const float* ln1g = (const float*)d_in[14];  const float* ln1b = (const float*)d_in[15];
    const float* qkv_w = (const float*)d_in[16];
    const float* proj_w = (const float*)d_in[17]; const float* proj_b = (const float*)d_in[18];
    const float* gam_w = (const float*)d_in[19];  const float* gam_b = (const float*)d_in[20];
    const float* ln2g = (const float*)d_in[21];   const float* ln2b = (const float*)d_in[22];
    const float* mlp_w1 = (const float*)d_in[23]; const float* mlp_b1 = (const float*)d_in[24];
    const float* mlp_w2 = (const float*)d_in[25]; const float* mlp_b2 = (const float*)d_in[26];
    const float* head_lng = (const float*)d_in[27]; const float* head_lnb = (const float*)d_in[28];
    const float* head_w = (const float*)d_in[29];   const float* head_b = (const float*)d_in[30];
    const float* prj_w1 = (const float*)d_in[31];   const float* prj_b1 = (const float*)d_in[32];
    const float* prj_w2 = (const float*)d_in[33];   const float* prj_b2 = (const float*)d_in[34];
    const float* prj_w3 = (const float*)d_in[35];   const float* prj_b3 = (const float*)d_in[36];
    const float* prj_lng = (const float*)d_in[37];  const float* prj_lnb = (const float*)d_in[38];
    const float* prj_w4 = (const float*)d_in[39];   const float* prj_b4 = (const float*)d_in[40];

    char* wsp = (char*)d_ws;
    size_t off = 0;
    auto take = [&](size_t bytes) -> void* {
        void* p = wsp + off;
        off += (bytes + 255) & ~((size_t)255);
        return p;
    };
    float*    xF    = (float*)take((size_t)TOK * DIMX * 4);
    float*    hF    = (float*)take((size_t)TOK * DIMX * 4);
    _Float16* h16   = (_Float16*)take((size_t)TOK * DIMX * 2);
    float*    gamma = (float*)take((size_t)TOK * 2 * 4);
    _Float16* Q16   = (_Float16*)take((size_t)64 * MM * DIMX * 2);
    _Float16* K16   = (_Float16*)take((size_t)64 * MM * DIMX * 2);
    _Float16* V16   = (_Float16*)take((size_t)64 * MM * DIMX * 2);
    _Float16* O16   = (_Float16*)take((size_t)TOK * 1024 * 2);
    _Float16* mlp16 = (_Float16*)take((size_t)TOK * 256 * 2);
    _Float16* wqkv  = (_Float16*)take((size_t)5 * 128 * 3072 * 2);
    _Float16* wproj = (_Float16*)take((size_t)5 * 1024 * 128 * 2);
    _Float16* wm1   = (_Float16*)take((size_t)5 * 128 * 256 * 2);
    _Float16* wm2   = (_Float16*)take((size_t)5 * 256 * 128 * 2);

    {
        int n;
        n = 5 * 128 * 3072; cvt_f32_f16<<<(n + 255) / 256, 256, 0, stream>>>(qkv_w, wqkv, n);
        n = 5 * 1024 * 128; cvt_f32_f16<<<(n + 255) / 256, 256, 0, stream>>>(proj_w, wproj, n);
        n = 5 * 128 * 256;  cvt_f32_f16<<<(n + 255) / 256, 256, 0, stream>>>(mlp_w1, wm1, n);
        n = 5 * 256 * 128;  cvt_f32_f16<<<(n + 255) / 256, 256, 0, stream>>>(mlp_w2, wm2, n);
    }
    pe_kernel<<<(NB * NN) / 256, 256, 0, stream>>>(node_feat, lapl, pe_w1, pe_b1, pe_w2, pe_b2,
                                                   pe_w3, pe_b3, pe_lng, pe_lnb, pe_w4, pe_b4, xF);
    init_cls<<<(NB * DIMX + 255) / 256, 256, 0, stream>>>(cls, xF);

    dim3 gLN((TOK + 7) / 8);
    for (int l = 0; l < 5; ++l) {
        ln_kernel<<<gLN, 256, 0, stream>>>(xF, ln1g + l * 128, ln1b + l * 128, hF, h16,
                                           gam_w + (size_t)l * 256, gam_b + l * 2, gamma, TOK);
        gemm_kernel<<<dim3((TOK + 63) / 64, 3072 / 64), 256, 0, stream>>>(
            h16, wqkv + (size_t)l * 128 * 3072, TOK, 128, 3072, 0,
            nullptr, nullptr, nullptr, nullptr, Q16, K16, V16);
        attn_kernel<<<dim3(64, 9), 256, 0, stream>>>(Q16, K16, V16, gamma, adj, O16);
        gemm_kernel<<<dim3((TOK + 63) / 64, 128 / 64), 256, 0, stream>>>(
            O16, wproj + (size_t)l * 1024 * 128, TOK, 1024, 128, 1,
            proj_b + l * 128, hF, xF, nullptr, nullptr, nullptr, nullptr);
        ln_kernel<<<gLN, 256, 0, stream>>>(xF, ln2g + l * 128, ln2b + l * 128, hF, h16,
                                           nullptr, nullptr, nullptr, TOK);
        gemm_kernel<<<dim3((TOK + 63) / 64, 256 / 64), 256, 0, stream>>>(
            h16, wm1 + (size_t)l * 128 * 256, TOK, 128, 256, 2,
            mlp_b1 + l * 256, nullptr, nullptr, mlp16, nullptr, nullptr, nullptr);
        gemm_kernel<<<dim3((TOK + 63) / 64, 128 / 64), 256, 0, stream>>>(
            mlp16, wm2 + (size_t)l * 256 * 128, TOK, 256, 128, 1,
            mlp_b2 + l * 128, hF, xF, nullptr, nullptr, nullptr, nullptr);
    }
    head_kernel<<<1, 256, 0, stream>>>(xF, head_lng, head_lnb, head_w, head_b,
                                       prj_w1, prj_b1, prj_w2, prj_b2, prj_w3, prj_b3,
                                       prj_lng, prj_lnb, prj_w4, prj_b4, (float*)d_out);
}
